// MyfAttention_62775241999201
// MI455X (gfx1250) — matrix operand order repaired and checked by reading, not compiled, not run on hardware
//
#include <hip/hip_runtime.h>
#include <hip/hip_bf16.h>

typedef __bf16 bf16;
typedef __bf16 v16bf __attribute__((ext_vector_type(16)));
typedef float  v8f   __attribute__((ext_vector_type(8)));

#define B_  2
#define S_  2048
#define H_  1024
#define NH_ 16
#define HD_ 64
#define BS_ (B_*S_)

union FragU { uint4 u[2]; v16bf v; };

__device__ inline v8f wmma_bf16(v16bf a, v16bf b, v8f c) {
  return __builtin_amdgcn_wmma_f32_16x16x32_bf16(false, a, false, b,
                                                 (short)0, c, false, false);
}

// A-fragment (16x32 bf16): lane = row, halves[0..15]=row[2*base..2*base+15],
// base = (lane>=16)?8:0.
__device__ inline v16bf load_a_frag(const bf16* row, int base) {
  FragU f;
  f.u[0] = *(const uint4*)(row + 2 * base);
  f.u[1] = *(const uint4*)(row + 2 * base + 8);
  return f.v;
}

// ---------------- elementwise f32 -> bf16 ----------------
__global__ void cvt_x_kernel(const float* __restrict__ X, bf16* __restrict__ Xb, int n) {
  int i = (blockIdx.x * blockDim.x + threadIdx.x) * 4;
  if (i < n) {
    float4 v = *(const float4*)(X + i);
    Xb[i + 0] = (bf16)v.x;
    Xb[i + 1] = (bf16)v.y;
    Xb[i + 2] = (bf16)v.z;
    Xb[i + 3] = (bf16)v.w;
  }
}

// ---------------- W[k][n] f32  ->  Wt[n][k] bf16 (LDS tiled) ----------------
__global__ void wt_kernel(const float* __restrict__ W, bf16* __restrict__ Wt) {
  __shared__ float tile[32][33];
  const int tx = threadIdx.x, ty = threadIdx.y;  // 32 x 8
  const int n0 = blockIdx.x * 32, k0 = blockIdx.y * 32;
#pragma unroll
  for (int r = 0; r < 4; ++r)
    tile[ty + 8 * r][tx] = W[(size_t)(k0 + ty + 8 * r) * H_ + n0 + tx];
  __syncthreads();
#pragma unroll
  for (int r = 0; r < 4; ++r)
    Wt[(size_t)(n0 + ty + 8 * r) * H_ + k0 + tx] = (bf16)tile[tx][ty + 8 * r];
}

// ---------------- generic bf16 GEMM: C[M,N] = A[M,K] * Wt[N,K]^T ------------
// Each wave: 32-row tile x 64 columns (2 m-subtiles x 4 WMMA n-tiles),
// K-loop step 32.  Each B-fragment feeds 2 WMMAs; the 4 waves of a block
// share the same n-group so the heavy B stream broadcasts out of L1.
// MODE 0: store bf16 row-major, scaled.  MODE 1: store bf16 as Vt[b][c][s].
// MODE 2: store f32 row-major.
template <int MODE>
__global__ void __launch_bounds__(128)
gemm16_kernel(const bf16* __restrict__ A, const bf16* __restrict__ Wt,
              void* __restrict__ outp, float scale) {
  const int lane = threadIdx.x & 31;
  const int wid  = threadIdx.x >> 5;
  const int flat = blockIdx.x * 4 + wid;   // 0..2047
  const int ng   = flat >> 7;              // 16 n-groups of 64 (same for whole block)
  const int mt   = flat & 127;             // 128 m-tiles of 32 rows (consecutive in block)
  const int hl   = lane & 15;
  const int hi   = (lane >> 4) & 1;

  const bf16* arow0 = A + (size_t)(mt * 32 + hl) * H_;
  const bf16* arow1 = arow0 + (size_t)16 * H_;
  const int   abase = hi * 8;
  const int   n0    = ng * 64 + hl;
  const int   koffB = hi * 16;

  v8f acc0[4] = {}, acc1[4] = {};
  for (int k0 = 0; k0 < H_; k0 += 32) {
    v16bf af0 = load_a_frag(arow0 + k0, abase);
    v16bf af1 = load_a_frag(arow1 + k0, abase);
    __builtin_prefetch(arow0 + k0 + 64, 0, 1);   // global_prefetch_b8
#pragma unroll
    for (int nt = 0; nt < 4; ++nt) {
      const bf16* bp = Wt + (size_t)(n0 + nt * 16) * H_ + k0 + koffB;
      v16bf bfrag = *(const v16bf*)bp;           // contiguous 32B per lane
      acc0[nt] = wmma_bf16(af0, bfrag, acc0[nt]);
      acc1[nt] = wmma_bf16(af1, bfrag, acc1[nt]);
    }
  }
#pragma unroll
  for (int part = 0; part < 2; ++part) {
#pragma unroll
    for (int nt = 0; nt < 4; ++nt) {
#pragma unroll
      for (int j = 0; j < 8; ++j) {
        const float v = (part == 0) ? acc0[nt][j] : acc1[nt][j];
        const int M   = j + hi * 8;
        const int row = mt * 32 + part * 16 + M;
        const int c   = ng * 64 + nt * 16 + hl;
        if (MODE == 0) {
          ((bf16*)outp)[(size_t)row * H_ + c] = (bf16)(v * scale);
        } else if (MODE == 1) {
          const int b = row / S_, s = row % S_;
          ((bf16*)outp)[((size_t)b * H_ + c) * S_ + s] = (bf16)v;
        } else {
          ((float*)outp)[(size_t)row * H_ + c] = v;
        }
      }
    }
  }
}

// ---------------- flash attention: wave = (b, h, 16-query tile) -------------
__global__ void __launch_bounds__(128)
attn_kernel(const bf16* __restrict__ Q, const bf16* __restrict__ Kb,
            const bf16* __restrict__ Vt, const int* __restrict__ mask,
            bf16* __restrict__ ctx) {
  __shared__ __align__(16) bf16 Pst[4][16][32];  // per-wave 16x32 P staging
  const int lane = threadIdx.x & 31;
  const int w    = threadIdx.x >> 5;
  const int flat = blockIdx.x * 4 + w;   // 0..4095
  const int qi   = flat & 127;           // S/16 tiles
  const int h    = (flat >> 7) & 15;
  const int b    = flat >> 11;
  const int q0   = qi * 16;
  const int hl   = lane & 15;
  const int hi   = (lane >> 4) & 1;

  // Q fragments for K-dim 0..31 and 32..63 (Q pre-scaled by HD^-0.5)
  const bf16* qrow = Q + (size_t)(b * S_ + q0 + hl) * H_ + h * HD_;
  const v16bf qf0 = load_a_frag(qrow, hi * 8);
  const v16bf qf1 = load_a_frag(qrow + 32, hi * 8);

  float m[8], l[8];
#pragma unroll
  for (int j = 0; j < 8; ++j) { m[j] = -1e30f; l[j] = 0.0f; }
  v8f O[4] = {};

  for (int kb = 0; kb < S_; kb += 32) {
    // scores: two 16x16 tiles (keys kb..+15 and kb+16..+31), K-dim = 64
    v8f s0 = {}, s1 = {};
    {
      const bf16* kr0 = Kb + (size_t)(b * S_ + kb + hl) * H_ + h * HD_ + hi * 16;
      const bf16* kr1 = Kb + (size_t)(b * S_ + kb + 16 + hl) * H_ + h * HD_ + hi * 16;
      s0 = wmma_bf16(qf0, *(const v16bf*)kr0, s0);
      s0 = wmma_bf16(qf1, *(const v16bf*)(kr0 + 32), s0);
      s1 = wmma_bf16(qf0, *(const v16bf*)kr1, s1);
      s1 = wmma_bf16(qf1, *(const v16bf*)(kr1 + 32), s1);
    }
    // clip + mask (uniform control flow: ternaries keep EXEC full)
    const int mk0 = mask[b * S_ + kb + hl];
    const int mk1 = mask[b * S_ + kb + 16 + hl];
#pragma unroll
    for (int j = 0; j < 8; ++j) {
      float a0 = fminf(fmaxf(s0[j], -10000.0f), 10000.0f);
      float a1 = fminf(fmaxf(s1[j], -10000.0f), 10000.0f);
      s0[j] = mk0 ? a0 : -1e30f;
      s1[j] = mk1 ? a1 : -1e30f;
    }
    // online softmax; rows live across 16-lane half-waves -> shfl_xor reduce
    float corr[8], p0[8], p1[8];
#pragma unroll
    for (int j = 0; j < 8; ++j) {
      float t = fmaxf(s0[j], s1[j]);
      t = fmaxf(t, __shfl_xor(t, 1));
      t = fmaxf(t, __shfl_xor(t, 2));
      t = fmaxf(t, __shfl_xor(t, 4));
      t = fmaxf(t, __shfl_xor(t, 8));
      const float mn = fmaxf(m[j], t);
      corr[j] = __expf(m[j] - mn);
      p0[j]   = __expf(s0[j] - mn);
      p1[j]   = __expf(s1[j] - mn);
      float rs = p0[j] + p1[j];
      rs += __shfl_xor(rs, 1);
      rs += __shfl_xor(rs, 2);
      rs += __shfl_xor(rs, 4);
      rs += __shfl_xor(rs, 8);
      l[j] = l[j] * corr[j] + rs;
      m[j] = mn;
    }
#pragma unroll
    for (int dt = 0; dt < 4; ++dt)
#pragma unroll
      for (int j = 0; j < 8; ++j) O[dt][j] *= corr[j];

    // C-layout P -> LDS (row-major 16x32) -> A-fragment reload
#pragma unroll
    for (int j = 0; j < 8; ++j) {
      const int r = j + hi * 8;
      Pst[w][r][hl]      = (bf16)p0[j];
      Pst[w][r][hl + 16] = (bf16)p1[j];
    }
    asm volatile("s_wait_dscnt 0x0" ::: "memory");  // same-wave DS in-order + fence
    v16bf pf;
    {
      FragU f;
      const bf16* pr = &Pst[w][hl][0];
      f.u[0] = *(const uint4*)(pr + hi * 16);
      f.u[1] = *(const uint4*)(pr + hi * 16 + 8);
      pf = f.v;
    }
    // O += P @ V  (V pre-transposed: Vt[(b*H + h*64 + d)*S + key])
#pragma unroll
    for (int dt = 0; dt < 4; ++dt) {
      const bf16* vp = Vt + ((size_t)b * H_ + h * HD_ + dt * 16 + hl) * S_ + kb + hi * 16;
      O[dt] = wmma_bf16(pf, *(const v16bf*)vp, O[dt]);
    }
    asm volatile("" ::: "memory");  // keep next iter's P stores after this iter's loads
  }

  float rl[8];
#pragma unroll
  for (int j = 0; j < 8; ++j) rl[j] = 1.0f / l[j];
#pragma unroll
  for (int dt = 0; dt < 4; ++dt)
#pragma unroll
    for (int j = 0; j < 8; ++j) {
      const int row = b * S_ + q0 + j + hi * 8;
      const int c   = h * HD_ + dt * 16 + hl;
      ctx[(size_t)row * H_ + c] = (bf16)(O[dt][j] * rl[j]);
    }
}

extern "C" void kernel_launch(void* const* d_in, const int* in_sizes, int n_in,
                              void* d_out, int out_size, void* d_ws, size_t ws_size,
                              hipStream_t stream) {
  (void)in_sizes; (void)n_in; (void)out_size; (void)ws_size;
  const float* X    = (const float*)d_in[0];
  const int*   mask = (const int*)d_in[1];
  const float* WQ   = (const float*)d_in[2];
  const float* WK   = (const float*)d_in[3];
  const float* WV   = (const float*)d_in[4];
  const float* WO   = (const float*)d_in[5];
  float* out = (float*)d_out;

  const size_t XB = (size_t)BS_ * H_ * sizeof(bf16);  // 8 MiB
  const size_t WB = (size_t)H_ * H_ * sizeof(bf16);   // 2 MiB
  char* ws = (char*)d_ws;
  bf16* Xb  = (bf16*)(ws);
  bf16* WtQ = (bf16*)(ws + XB);
  bf16* WtK = (bf16*)(ws + XB + WB);
  bf16* WtV = (bf16*)(ws + XB + 2 * WB);
  bf16* WtO = (bf16*)(ws + XB + 3 * WB);
  bf16* Qb  = (bf16*)(ws + XB + 4 * WB);
  bf16* Kb  = (bf16*)(ws + XB + 4 * WB + XB);
  bf16* Vt  = (bf16*)(ws + XB + 4 * WB + 2 * XB);
  bf16* Ctx = (bf16*)(ws + XB + 4 * WB + 3 * XB);

  const int nX = BS_ * H_;
  cvt_x_kernel<<<(nX / 4 + 255) / 256, 256, 0, stream>>>(X, Xb, nX);

  dim3 tb(32, 8), tg(32, 32);
  wt_kernel<<<tg, tb, 0, stream>>>(WQ, WtQ);
  wt_kernel<<<tg, tb, 0, stream>>>(WK, WtK);
  wt_kernel<<<tg, tb, 0, stream>>>(WV, WtV);
  wt_kernel<<<tg, tb, 0, stream>>>(WO, WtO);

  const float scaling = 0.125f;  // HD^-0.5
  gemm16_kernel<0><<<512, 128, 0, stream>>>(Xb, WtQ, Qb, scaling);
  gemm16_kernel<0><<<512, 128, 0, stream>>>(Xb, WtK, Kb, 1.0f);
  gemm16_kernel<1><<<512, 128, 0, stream>>>(Xb, WtV, Vt, 1.0f);

  attn_kernel<<<1024, 128, 0, stream>>>(Qb, Kb, Vt, mask, Ctx);

  gemm16_kernel<2><<<512, 128, 0, stream>>>(Ctx, WtO, out, 1.0f);
}
